// HeteroGraphSAGE_78228534329621
// MI455X (gfx1250) — compile-verified, matrix operand order checked by reading, not run on hardware
//
#include <hip/hip_runtime.h>
#include <cstdint>
#include <cstddef>

// ---------------------------------------------------------------------------
// HeteroGraphSAGE-style GPR-GNN forward for MI455X (gfx1250, wave32, WMMA).
// - Dense matmuls: v_wmma_f32_16x16x32_f16 (f32 -> f16 fragments, f32 accum).
// - ADJ_P = hidden@hidden^T: 5-wide WMMA SYRK with non-temporal (TH=NT) stores
//   so the 576 MB streaming output does not evict the <64 MB L2 working set.
// - Graph convs: edge-parallel gather + global f32 atomics, with
//   global_prefetch lookahead on the data-dependent gather rows.
// - Attention-score kernel stages fcW (16 KB) into LDS with
//   global_load_async_to_lds_b128 (ASYNCcnt) + s_wait_asynccnt.
// - Loop-invariant conv (m_t0 from fixed Tx1_o) hoisted; dead m_o in the loop
//   skipped; loss_dis1 == 0 analytically (1-row readout => centered X = 0).
// - Launch-count minimized: fused eta+axpy (wave32 shfl reduce), combine
//   kernel also re-zeroes the scatter accumulator for the next iteration.
// ---------------------------------------------------------------------------

typedef __attribute__((ext_vector_type(16))) _Float16 v16h;
typedef __attribute__((ext_vector_type(8)))  float    v8f;

#define NT   12000
#define NO   24000
#define HIDD 64
#define RNK  32
#define KPROP 10
#define TCOLS 5   // SYRK column tiles per wave; 750 tiles = 5 * 150

// WMMA 16x16x32 f16 A/B fragment K-index for element j of a lane's v16h.
__device__ __forceinline__ int kmap16(int j, int half) {
  return ((j < 8) ? j : j + 8) + half * 8;
}

// ---------------- generic WMMA GEMM: C = act(rowscale(A) @ B + bias) --------
// A: MxK f32 row-major, B: KxN f32 row-major, C: MxN f32. M assumed %16==0.
// ACT: 0=none, 1=tanh, 2=relu. KTAIL: guard k<K (K%32!=0). NGUARD: guard
// col<N (N%16!=0). NTST: non-temporal stores.
template <int ACT, bool KTAIL, bool NGUARD, bool NTST>
__global__ void wmma_gemm_kernel(const float* __restrict__ A,
                                 const float* __restrict__ B,
                                 const float* __restrict__ bias,
                                 const float* __restrict__ rowscale,
                                 float* __restrict__ C,
                                 int M, int N, int K, int ldc) {
  const int lane = threadIdx.x & 31;
  const int m    = lane & 15;
  const int half = lane >> 4;
  const int row  = blockIdx.y * 16 + m;
  const int col  = blockIdx.x * 16 + m;
  const float rs = (rowscale != nullptr) ? rowscale[row] : 1.0f;

  v8f acc = {};
  for (int k0 = 0; k0 < K; k0 += 32) {
    v16h a, b;
#pragma unroll
    for (int j = 0; j < 16; ++j) {
      const int k = k0 + kmap16(j, half);
      float av, bv;
      if (KTAIL || NGUARD) {
        av = (!KTAIL || k < K) ? A[(size_t)row * K + k] : 0.0f;
        bv = ((!KTAIL || k < K) && (!NGUARD || col < N))
                 ? B[(size_t)k * N + col] : 0.0f;
      } else {
        av = A[(size_t)row * K + k];
        bv = B[(size_t)k * N + col];
      }
      a[j] = (_Float16)(av * rs);
      b[j] = (_Float16)bv;
    }
    acc = __builtin_amdgcn_wmma_f32_16x16x32_f16(false, a, false, b,
                                                 (short)0, acc, false, false);
  }
  const float bval = (bias != nullptr && (!NGUARD || col < N)) ? bias[col] : 0.0f;
#pragma unroll
  for (int r = 0; r < 8; ++r) {
    const int orow = blockIdx.y * 16 + r + half * 8;
    if (!NGUARD || col < N) {
      float v = acc[r] + bval;
      if (ACT == 1) v = tanhf(v);
      if (ACT == 2) v = fmaxf(v, 0.0f);
      if (NTST) __builtin_nontemporal_store(v, &C[(size_t)orow * ldc + col]);
      else      C[(size_t)orow * ldc + col] = v;
    }
  }
}

template <int ACT, bool KTAIL, bool NGUARD, bool NTST>
static void launch_gemm(const float* A, const float* B, const float* bias,
                        const float* rsc, float* C, int M, int N, int K,
                        hipStream_t s) {
  dim3 g((N + 15) / 16, (M + 15) / 16);
  wmma_gemm_kernel<ACT, KTAIL, NGUARD, NTST><<<g, 32, 0, s>>>(
      A, B, bias, rsc, C, M, N, K, N);
}

// ---------------- WMMA SYRK: C = A16 @ A16^T, A16: n x 64 f16 ---------------
// One wave computes 16 x (16*TCOLS); A fragments reused across TCOLS tiles.
// Output streamed with non-temporal stores (write-once 576 MB).
__global__ void wmma_syrk_kernel(const _Float16* __restrict__ A16,
                                 float* __restrict__ C, int n) {
  const int lane = threadIdx.x & 31;
  const int m    = lane & 15;
  const int half = lane >> 4;
  const int row  = blockIdx.y * 16 + m;
  const int colbase = blockIdx.x * (16 * TCOLS);

  v16h a0, a1;
#pragma unroll
  for (int j = 0; j < 16; ++j) {
    a0[j] = A16[(size_t)row * 64 + kmap16(j, half)];
    a1[j] = A16[(size_t)row * 64 + 32 + kmap16(j, half)];
  }
  v8f acc[TCOLS];
#pragma unroll
  for (int t = 0; t < TCOLS; ++t) {
    const int col = colbase + t * 16 + m;
    v16h b0, b1;
#pragma unroll
    for (int j = 0; j < 16; ++j) {
      b0[j] = A16[(size_t)col * 64 + kmap16(j, half)];
      b1[j] = A16[(size_t)col * 64 + 32 + kmap16(j, half)];
    }
    v8f c = {};
    c = __builtin_amdgcn_wmma_f32_16x16x32_f16(false, a0, false, b0,
                                               (short)0, c, false, false);
    c = __builtin_amdgcn_wmma_f32_16x16x32_f16(false, a1, false, b1,
                                               (short)0, c, false, false);
    acc[t] = c;
  }
#pragma unroll
  for (int r = 0; r < 8; ++r) {
    const size_t orow = (size_t)(blockIdx.y * 16 + r + half * 8) * n;
#pragma unroll
    for (int t = 0; t < TCOLS; ++t) {
      __builtin_nontemporal_store(acc[t][r], &C[orow + colbase + t * 16 + m]);
    }
  }
}

// ---------------- small utility kernels ------------------------------------
__global__ void fill_kernel(float* __restrict__ p, int n) {
  int i = blockIdx.x * blockDim.x + threadIdx.x;
  if (i < n) p[i] = 0.0f;
}

__global__ void deg_count_kernel(const int* __restrict__ idx, int e,
                                 float* __restrict__ deg) {
  int i = blockIdx.x * blockDim.x + threadIdx.x;
  if (i < e) atomicAdd(&deg[idx[i]], 1.0f);
}

__global__ void deg_rsqrt_kernel(float* __restrict__ deg, int n) {
  int i = blockIdx.x * blockDim.x + threadIdx.x;
  if (i < n) deg[i] = rsqrtf(fmaxf(deg[i], 1.0f));
}

// edge-parallel gather (scaled by D_src^-1/2) + scatter-add. 16 lanes/edge,
// float4 per lane -> 64 feats. Lookahead prefetch of the gathered row.
__global__ void scatter_gc_kernel(const float* __restrict__ x,
                                  const int* __restrict__ src,
                                  const int* __restrict__ dst,
                                  const float* __restrict__ rs_src,
                                  float* __restrict__ agg, int e) {
  int tid = blockIdx.x * blockDim.x + threadIdx.x;
  int ed = tid >> 4;
  int q  = tid & 15;
  if (ed >= e) return;
  int s = src[ed], d = dst[ed];
  if (q == 0) {
    int ep = ed + 4096;
    if (ep < e) __builtin_prefetch(&x[(size_t)src[ep] * HIDD], 0, 0);
  }
  float sc = rs_src[s];
  float4 v = reinterpret_cast<const float4*>(x + (size_t)s * HIDD)[q];
  float* base = agg + (size_t)d * HIDD + q * 4;
  atomicAdd(base + 0, v.x * sc);
  atomicAdd(base + 1, v.y * sc);
  atomicAdd(base + 2, v.z * sc);
  atomicAdd(base + 3, v.w * sc);
}

// attention score: slot += (1/N) * sum_rows v . tanh(row @ fcW + fcb)
// fcW (64x64 f32 = 16 KB) staged into LDS via gfx1250 async DMA loads:
// global_load_async_to_lds_b128 (ASYNCcnt), completed with s_wait_asynccnt.
__global__ void att_score_kernel(const float* __restrict__ E, int nrows,
                                 const float* __restrict__ fcW,
                                 const float* __restrict__ fcb,
                                 const float* __restrict__ vvec,
                                 float* __restrict__ slot) {
  __shared__ float sW[HIDD * HIDD];
  __shared__ float sb[HIDD];
  __shared__ float sv[HIDD];
  {
    // LDS aperture: low 32 bits of a flat shared pointer are the LDS offset.
    uint32_t lds_base = (uint32_t)(uintptr_t)(void*)sW;
#pragma unroll
    for (int c = 0; c < 4; ++c) {
      uint32_t byte_off = (threadIdx.x + c * 256u) * 16u;   // 4 KB per round
      uint32_t lds_addr = lds_base + byte_off;
      uint64_t gaddr = (uint64_t)(uintptr_t)fcW + byte_off;
      asm volatile("global_load_async_to_lds_b128 %0, %1, off"
                   :: "v"(lds_addr), "v"(gaddr)
                   : "memory");
    }
    for (int i = threadIdx.x; i < HIDD; i += blockDim.x) { sb[i] = fcb[i]; sv[i] = vvec[i]; }
    asm volatile("s_wait_asynccnt 0x0" ::: "memory");
  }
  __syncthreads();
  int row = blockIdx.x * blockDim.x + threadIdx.x;
  if (row < nrows) {
    float e[HIDD];
#pragma unroll
    for (int d = 0; d < HIDD; ++d) e[d] = E[(size_t)row * HIDD + d];
    float s = 0.0f;
    for (int h = 0; h < HIDD; ++h) {
      float acc = sb[h];
#pragma unroll
      for (int d = 0; d < HIDD; ++d) acc += e[d] * sW[d * HIDD + h];
      s += sv[h] * tanhf(acc);
    }
    atomicAdd(slot, s / (float)nrows);
  }
}

// out = scale*(b0*m0 + b1*m1) - (sub ? sub[i] : 0), beta = softmax(scores);
// optionally re-zeroes zbuf (the scatter accumulator) for the next iteration.
__global__ void combine_kernel(float* __restrict__ out,
                               const float* __restrict__ m0,
                               const float* __restrict__ m1,
                               const float* __restrict__ scores, int ia, int ib,
                               const float* __restrict__ sub, float scale,
                               float* __restrict__ zbuf, int n) {
  int i = blockIdx.x * blockDim.x + threadIdx.x;
  if (i >= n) return;
  float sa = scores[ia], sb = scores[ib];
  float mx = fmaxf(sa, sb);
  float ea = expf(sa - mx), eb = expf(sb - mx);
  float inv = 1.0f / (ea + eb);
  float v = (ea * inv) * m0[i] + (eb * inv) * m1[i];
  v = scale * v - (sub ? sub[i] : 0.0f);
  out[i] = v;
  if (zbuf) zbuf[i] = 0.0f;
}

// fused: eta = (hk[row,:] . gamma[:,kcol])/RANK (wave32 shfl reduce),
// then hidden[row,:] (+)= Tx[row,:] * eta. One 64-thread block per row.
__global__ void eta_axpy_kernel(const float* __restrict__ hk,
                                const float* __restrict__ gamma, int kcol,
                                const float* __restrict__ Tx,
                                float* __restrict__ hidden, int init) {
  const int row = blockIdx.x;
  const int t = threadIdx.x;  // 0..63 (2 waves)
  __shared__ float se;
  if (t < 32) {
    float p = hk[(size_t)row * RNK + t] * gamma[t * 11 + kcol];
    p += __shfl_xor(p, 16);
    p += __shfl_xor(p, 8);
    p += __shfl_xor(p, 4);
    p += __shfl_xor(p, 2);
    p += __shfl_xor(p, 1);
    if (t == 0) se = p * (1.0f / (float)RNK);
  }
  __syncthreads();
  const float e = se;
  const size_t i = (size_t)row * HIDD + t;
  float v = Tx[i] * e;
  hidden[i] = init ? v : (hidden[i] + v);
}

// hf32/out_h = tanh(hidden); a16 = (f16)hidden (pre-tanh, feeds ADJ syrk)
__global__ void tanh_convert_kernel(const float* __restrict__ hidden,
                                    float* __restrict__ hf32,
                                    float* __restrict__ out_h,
                                    _Float16* __restrict__ a16, int n) {
  int i = blockIdx.x * blockDim.x + threadIdx.x;
  if (i >= n) return;
  float v = hidden[i];
  float t = tanhf(v);
  hf32[i] = t;
  out_h[i] = t;
  a16[i] = (_Float16)v;
}

__global__ void logits_kernel(const float* __restrict__ h,
                              const float* __restrict__ W,
                              const float* __restrict__ b,
                              float* __restrict__ out, int n) {
  int row = blockIdx.x * blockDim.x + threadIdx.x;
  if (row >= n) return;
  float a0 = b[0], a1 = b[1], a2 = b[2];
#pragma unroll
  for (int d = 0; d < HIDD; ++d) {
    float hv = h[(size_t)row * HIDD + d];
    a0 += hv * W[d * 3 + 0];
    a1 += hv * W[d * 3 + 1];
    a2 += hv * W[d * 3 + 2];
  }
  out[(size_t)row * 3 + 0] = a0;
  out[(size_t)row * 3 + 1] = a1;
  out[(size_t)row * 3 + 2] = a2;
}

// slot += mean_feat((h-rowmean(h))*(zp-rowmean(zp))) per row
__global__ void cov_kernel(const float* __restrict__ h,
                           const float* __restrict__ zp,
                           float* __restrict__ slot, int n) {
  int row = blockIdx.x * blockDim.x + threadIdx.x;
  if (row >= n) return;
  float mh = 0.0f, mz = 0.0f;
#pragma unroll
  for (int d = 0; d < HIDD; ++d) { mh += h[(size_t)row * HIDD + d]; mz += zp[(size_t)row * HIDD + d]; }
  mh *= (1.0f / HIDD); mz *= (1.0f / HIDD);
  float c = 0.0f;
#pragma unroll
  for (int d = 0; d < HIDD; ++d)
    c += (h[(size_t)row * HIDD + d] - mh) * (zp[(size_t)row * HIDD + d] - mz);
  atomicAdd(slot, c * (1.0f / HIDD));
}

__global__ void loss_finalize_kernel(const float* __restrict__ slot,
                                     float* __restrict__ out) {
  float s = slot[0] * (1.0f / (float)NT);
  out[0] = s * s;   // loss_dis; loss_dis1 == 0 analytically
}

// ---------------------------------------------------------------------------
extern "C" void kernel_launch(void* const* d_in, const int* in_sizes, int n_in,
                              void* d_out, int out_size, void* d_ws, size_t ws_size,
                              hipStream_t stream) {
  const float* features   = (const float*)d_in[0];
  const float* feat_other = (const float*)d_in[1];
  const float* z_pre      = (const float*)d_in[2];
  const float* Wt = (const float*)d_in[3];  const float* bt = (const float*)d_in[4];
  const float* Wo = (const float*)d_in[5];  const float* bo = (const float*)d_in[6];
  const float* W1_r0 = (const float*)d_in[7];  const float* b1_r0 = (const float*)d_in[8];
  const float* W1_r1 = (const float*)d_in[9];  const float* b1_r1 = (const float*)d_in[10];
  const float* W1_r2 = (const float*)d_in[11]; const float* b1_r2 = (const float*)d_in[12];
  const float* att1_fcW = (const float*)d_in[13];
  const float* att1_fcb = (const float*)d_in[14];
  const float* att1_v   = (const float*)d_in[15];
  const float* W2_r0 = (const float*)d_in[16]; const float* b2_r0 = (const float*)d_in[17];
  const float* W2_r2 = (const float*)d_in[20]; const float* b2_r2 = (const float*)d_in[21];
  const float* att2_fcW = (const float*)d_in[22];
  const float* att2_fcb = (const float*)d_in[23];
  const float* att2_v   = (const float*)d_in[24];
  const float* projW = (const float*)d_in[25];
  const float* projb = (const float*)d_in[26];
  const float* gamma = (const float*)d_in[27];
  const float* p3_W = (const float*)d_in[28]; const float* p3_b = (const float*)d_in[29];
  const float* p1_W1 = (const float*)d_in[30]; const float* p1_b1 = (const float*)d_in[31];
  const float* p1_W2 = (const float*)d_in[32]; const float* p1_b2 = (const float*)d_in[33];
  const float* out_W = (const float*)d_in[34]; const float* out_b = (const float*)d_in[35];
  const int* src0 = (const int*)d_in[36]; const int* dst0 = (const int*)d_in[37];
  const int* src1 = (const int*)d_in[38]; const int* dst1 = (const int*)d_in[39];
  const int* src2 = (const int*)d_in[40]; const int* dst2 = (const int*)d_in[41];
  const int E0 = in_sizes[36], E1 = in_sizes[38], E2 = in_sizes[40];

  // ---- output layout (return-order concatenation) ----
  float* out = (float*)d_out;
  float* o_logits = out;                         // 12000*3
  float* o_h      = o_logits + NT * 3;           // 12000*64
  float* o_xpre   = o_h + NT * HIDD;             // 12000*334
  float* o_zpre1  = o_xpre + (size_t)NT * 334;   // 12000*334
  float* o_adj    = o_zpre1 + (size_t)NT * 334;  // 12000*12000
  float* o_loss   = o_adj + (size_t)NT * NT;     // 1

  // ---- workspace bump allocator ----
  uintptr_t wp = (uintptr_t)d_ws;
  auto walloc = [&](size_t nbytes) -> void* {
    uintptr_t p = (wp + 255) & ~(uintptr_t)255;
    wp = p + nbytes;
    return (void*)p;
  };
  // one contiguous degree block: rs_s0|rs_d0|rs_s1|rs_d1|rs_s2|rs_d2
  const int NDEG = 2 * NO + 4 * NT;
  float* degbuf = (float*)walloc((size_t)NDEG * 4);
  float* rs_s0 = degbuf;
  float* rs_d0 = rs_s0 + NO;
  float* rs_s1 = rs_d0 + NT;
  float* rs_d1 = rs_s1 + NT;
  float* rs_s2 = rs_d1 + NO;
  float* rs_d2 = rs_s2 + NT;
  float* TxA   = (float*)walloc((size_t)NT * HIDD * 4);   // Chebyshev T_{k-1}
  float* TxB   = (float*)walloc((size_t)NT * HIDD * 4);   // T_k
  float* TxC   = (float*)walloc((size_t)NT * HIDD * 4);   // T_{k+1} temp
  float* Tx0_o = (float*)walloc((size_t)NO * HIDD * 4);
  float* Tx1_o = (float*)walloc((size_t)NO * HIDD * 4);
  float* m0    = (float*)walloc((size_t)NT * HIDD * 4);   // m_t0 / m_t0-loop
  float* m2    = (float*)walloc((size_t)NT * HIDD * 4);   // m_t2
  float* aggT  = (float*)walloc((size_t)NT * HIDD * 4);
  float* aggO  = (float*)walloc((size_t)NO * HIDD * 4);
  float* hk    = (float*)walloc((size_t)NT * RNK * 4);
  float* hidden= (float*)walloc((size_t)NT * HIDD * 4);
  float* hf32  = (float*)walloc((size_t)NT * HIDD * 4);
  _Float16* a16= (_Float16*)walloc((size_t)NT * HIDD * 2);
  float* zp    = (float*)walloc((size_t)NT * HIDD * 4);
  float* t1    = (float*)walloc((size_t)NT * 128 * 4);
  float* scores= (float*)walloc(16 * 4);   // 0,1: layer1; 2: loop m0; 3..11: k; 12: cov

  auto fill = [&](float* p, int n) {
    fill_kernel<<<dim3((n + 255) / 256), dim3(256), 0, stream>>>(p, n);
  };
  auto scatter = [&](const float* x, const int* s, const int* d,
                     const float* rs, float* agg, int e) {
    int threads = e * 16;
    scatter_gc_kernel<<<dim3((threads + 255) / 256), dim3(256), 0, stream>>>(x, s, d, rs, agg, e);
  };
  auto score = [&](const float* Emat, const float* fcW, const float* fcb,
                   const float* v, int slot) {
    att_score_kernel<<<dim3((NT + 255) / 256), dim3(256), 0, stream>>>(
        Emat, NT, fcW, fcb, v, scores + slot);
  };
  auto eta_axpy = [&](const float* hkp, int kcol, const float* Tx, int init) {
    eta_axpy_kernel<<<dim3(NT), dim3(64), 0, stream>>>(hkp, gamma, kcol, Tx, hidden, init);
  };
  const int nTH = NT * HIDD, nOH = NO * HIDD;
  auto combine = [&](float* outp, const float* a, const float* b, int ia, int ib,
                     const float* sub, float scale, float* zbuf) {
    combine_kernel<<<dim3((nTH + 255) / 256), 256, 0, stream>>>(
        outp, a, b, scores, ia, ib, sub, scale, zbuf, nTH);
  };

  // ---- degrees (clip>=1, rsqrt) ----
  fill(scores, 16);
  fill(degbuf, NDEG);
  deg_count_kernel<<<dim3((E0 + 255) / 256), 256, 0, stream>>>(src0, E0, rs_s0);
  deg_count_kernel<<<dim3((E0 + 255) / 256), 256, 0, stream>>>(dst0, E0, rs_d0);
  deg_count_kernel<<<dim3((E1 + 255) / 256), 256, 0, stream>>>(src1, E1, rs_s1);
  deg_count_kernel<<<dim3((E1 + 255) / 256), 256, 0, stream>>>(dst1, E1, rs_d1);
  deg_count_kernel<<<dim3((E2 + 255) / 256), 256, 0, stream>>>(src2, E2, rs_s2);
  deg_count_kernel<<<dim3((E2 + 255) / 256), 256, 0, stream>>>(dst2, E2, rs_d2);
  deg_rsqrt_kernel<<<dim3((NDEG + 255) / 256), 256, 0, stream>>>(degbuf, NDEG);

  // ---- input projections ----
  launch_gemm<0, true,  false, false>(features,   Wt, bt, nullptr, TxA,   NT, HIDD, 334, stream);
  launch_gemm<0, false, false, false>(feat_other, Wo, bo, nullptr, Tx0_o, NO, HIDD, 128, stream);

  // ---- h0/eta0, hidden init ----
  launch_gemm<1, false, false, false>(TxA, projW, projb, nullptr, hk, NT, RNK, HIDD, stream);
  eta_axpy(hk, 0, TxA, 1);

  // ---- layer-1 hetero conv ----
  fill(aggT, nTH);
  scatter(Tx0_o, src0, dst0, rs_s0, aggT, E0);
  launch_gemm<0, false, false, false>(aggT, W1_r0, b1_r0, rs_d0, m0, NT, HIDD, HIDD, stream);
  fill(aggT, nTH);
  scatter(TxA, src2, dst2, rs_s2, aggT, E2);
  launch_gemm<0, false, false, false>(aggT, W1_r2, b1_r2, rs_d2, m2, NT, HIDD, HIDD, stream);
  fill(aggO, nOH);
  scatter(TxA, src1, dst1, rs_s1, aggO, E1);
  launch_gemm<0, false, false, false>(aggO, W1_r1, b1_r1, rs_d1, Tx1_o, NO, HIDD, HIDD, stream);
  score(m0, att1_fcW, att1_fcb, att1_v, 0);
  score(m2, att1_fcW, att1_fcb, att1_v, 1);
  combine(TxB, m0, m2, 0, 1, nullptr, 1.0f, aggT);  // Tx1_t; re-zero aggT

  // ---- h1/eta1 ----
  launch_gemm<1, false, false, false>(TxB, projW + 1 * HIDD * RNK, projb + RNK,
                                      nullptr, hk, NT, RNK, HIDD, stream);
  eta_axpy(hk, 1, TxB, 0);

  // ---- loop-invariant: m_t0-loop from fixed Tx1_o (conv2, relation r0) ----
  scatter(Tx1_o, src0, dst0, rs_s0, aggT, E0);
  launch_gemm<0, false, false, false>(aggT, W2_r0, b2_r0, rs_d0, m0, NT, HIDD, HIDD, stream);
  score(m0, att2_fcW, att2_fcb, att2_v, 2);
  fill(aggT, nTH);

  // ---- GPR Chebyshev propagation (K-1 = 9 iterations) ----
  float* Tx0p = TxA; float* Tx1p = TxB; float* Tx2p = TxC;
  for (int k = 1; k < KPROP; ++k) {
    scatter(Tx1p, src2, dst2, rs_s2, aggT, E2);
    launch_gemm<0, false, false, false>(aggT, W2_r2, b2_r2, rs_d2, m2, NT, HIDD, HIDD, stream);
    score(m2, att2_fcW, att2_fcb, att2_v, 3 + (k - 1));
    // Tx2 = 2*(b0*m0 + b1*m2) - Tx0; also re-zeroes aggT for next iteration
    combine(Tx2p, m0, m2, 2, 3 + (k - 1), Tx0p, 2.0f, aggT);
    float* tmp = Tx0p; Tx0p = Tx1p; Tx1p = Tx2p; Tx2p = tmp;
    launch_gemm<1, false, false, false>(Tx1p, projW + (size_t)(k + 1) * HIDD * RNK,
                                        projb + (k + 1) * RNK, nullptr, hk,
                                        NT, RNK, HIDD, stream);
    eta_axpy(hk, k + 1, Tx1p, 0);
  }

  // ---- finals ----
  tanh_convert_kernel<<<dim3((nTH + 255) / 256), 256, 0, stream>>>(hidden, hf32, o_h, a16, nTH);
  // ADJ_P = hidden @ hidden^T  (150 x 750 blocks of 16x80 WMMA tiles)
  wmma_syrk_kernel<<<dim3(NT / (16 * TCOLS), NT / 16), 32, 0, stream>>>(a16, o_adj, NT);
  // z_p = z_pre @ p3_W + p3_b
  launch_gemm<0, false, false, false>(z_pre, p3_W, p3_b, nullptr, zp, NT, HIDD, 128, stream);
  // x_pre = tanh(relu(h @ p1_W1 + b1) @ p1_W2 + b2)   (streamed NT stores)
  launch_gemm<2, false, false, false>(hf32, p1_W1, p1_b1, nullptr, t1, NT, 128, HIDD, stream);
  launch_gemm<1, false, true,  true >(t1, p1_W2, p1_b2, nullptr, o_xpre, NT, 334, 128, stream);
  // z_pre1 = same MLP on z_p
  launch_gemm<2, false, false, false>(zp, p1_W1, p1_b1, nullptr, t1, NT, 128, HIDD, stream);
  launch_gemm<1, false, true,  true >(t1, p1_W2, p1_b2, nullptr, o_zpre1, NT, 334, 128, stream);
  // logits
  logits_kernel<<<dim3((NT + 255) / 256), 256, 0, stream>>>(hf32, out_W, out_b, o_logits, NT);
  // loss = cov(h, z_p)^2 ; hsic term is identically zero
  cov_kernel<<<dim3((NT + 255) / 256), 256, 0, stream>>>(hf32, zp, scores + 12, NT);
  loss_finalize_kernel<<<1, 1, 0, stream>>>(scores + 12, o_loss);
}
